// StyleLossPass2_23081154249116
// MI455X (gfx1250) — compile-verified, matrix operand order checked by reading, not run on hardware
//
#include <hip/hip_runtime.h>
#include <hip/hip_bf16.h>

// ---------------------------------------------------------------------------
// StyleLossPass2 for MI455X (gfx1250, wave32, WMMA).
//
// C=512, H=W=64, HW=4096, K9C=4608.
// Score GEMM cp(4096x4608) @ sp^T : ~155 GFLOP bf16 -> v_wmma_f32_16x16x32_bf16.
// Both bf16 patch matrices (75 MB) fit in 192 MB L2 -> matrix-pipe bound.
// Score GEMM: 128x128 block tile, 8 waves (2x4), 8 WMMAs/wave/K-step,
// double-buffered LDS fed by GLOBAL_LOAD_ASYNC_TO_LDS_B128 (ASYNCcnt).
// Argmax fused via packed u64 atomicMax. Patch extraction goes through an
// LDS transpose so both global sides are coalesced.
// ---------------------------------------------------------------------------

typedef __attribute__((ext_vector_type(16))) __bf16 v16bf;
typedef __attribute__((ext_vector_type(8)))  float  v8f;

// exact parameter types of the async-LDS builtin (from compiler diagnostic):
typedef int v4i_vec __attribute__((vector_size(16)));
typedef __attribute__((address_space(1))) v4i_vec as1_v4i;
typedef __attribute__((address_space(3))) v4i_vec as3_v4i;

#define CCH   512
#define HDIM  64
#define HW    4096
#define K9C   4608
#define EPSF  1e-9f

// ----- async global->LDS copy (CDNA5), with portable fallback ---------------
#if defined(__HIP_DEVICE_COMPILE__) && __has_builtin(__builtin_amdgcn_global_load_async_to_lds_b128)
#define SL2_ASYNC_LDS 1
__device__ __forceinline__ void sl2_copy_b128(const unsigned short* g, unsigned short* l) {
    __builtin_amdgcn_global_load_async_to_lds_b128((as1_v4i*)g, (as3_v4i*)l, 0, 0);
}
__device__ __forceinline__ void sl2_wait_async_4() {
#if __has_builtin(__builtin_amdgcn_s_wait_asynccnt)
    __builtin_amdgcn_s_wait_asynccnt(4);
#else
    asm volatile("s_wait_asynccnt 4" ::: "memory");
#endif
}
__device__ __forceinline__ void sl2_wait_async_0() {
#if __has_builtin(__builtin_amdgcn_s_wait_asynccnt)
    __builtin_amdgcn_s_wait_asynccnt(0);
#else
    asm volatile("s_wait_asynccnt 0" ::: "memory");
#endif
}
#else
#define SL2_ASYNC_LDS 0
__device__ __forceinline__ void sl2_copy_b128(const unsigned short* g, unsigned short* l) {
    *(uint4*)l = *(const uint4*)g;
}
__device__ __forceinline__ void sl2_wait_async_4() {}
__device__ __forceinline__ void sl2_wait_async_0() {}
#endif

__device__ __forceinline__ unsigned short f32_to_bf16_bits(float f) {
    unsigned int u = __float_as_uint(f);
    u += 0x7FFFu + ((u >> 16) & 1u);          // round to nearest even
    return (unsigned short)(u >> 16);
}
__device__ __forceinline__ float bf16_bits_to_f32(unsigned short s) {
    return __uint_as_float(((unsigned int)s) << 16);
}
__device__ __forceinline__ int reflect64(int x) {
    return x < 0 ? -x : (x > 63 ? 126 - x : x);
}

// ---------------------------------------------------------------------------
// 1) Patch extraction via LDS transpose: fm (C,H,W) f32 -> pm (HW, 9C) bf16.
//    Block = (shift s, 32-channel tile, 32-pixel tile).
//    Reads coalesced along pixels, writes coalesced along k (= s*512 + c).
// ---------------------------------------------------------------------------
__global__ __launch_bounds__(256) void sl2_patches_kernel(
    const float* __restrict__ fm, unsigned short* __restrict__ pm) {
    __shared__ float tile[32][33];
    const int bid = blockIdx.x;            // 9 * 16 * 128 blocks
    const int s   = bid / (16 * 128);
    const int rem = bid % (16 * 128);
    const int ct  = rem >> 7;              // 0..15  channel tile
    const int pt  = rem & 127;             // 0..127 pixel tile
    const int dh  = (s / 3) - 1;
    const int dw  = (s % 3) - 1;
    const int c0  = ct << 5;
    const int p0  = pt << 5;
    const int tid = threadIdx.x;

    #pragma unroll
    for (int pass = 0; pass < 4; ++pass) {
        const int cc = (tid >> 5) + (pass << 3);   // channel in tile
        const int i  = tid & 31;                   // pixel in tile (lane) -> coalesced
        const int p  = p0 + i;
        const int ri = reflect64((p >> 6) + dh);
        const int rj = reflect64((p & 63) + dw);
        tile[cc][i] = fm[(size_t)(c0 + cc) * HW + ri * HDIM + rj];
    }
    __syncthreads();
    #pragma unroll
    for (int pass = 0; pass < 4; ++pass) {
        const int i  = (tid >> 5) + (pass << 3);   // pixel
        const int cc = tid & 31;                   // channel (lane) -> coalesced writes
        pm[(size_t)(p0 + i) * K9C + (s << 9) + c0 + cc] = f32_to_bf16_bits(tile[cc][i]);
    }
}

// ---------------------------------------------------------------------------
// 2) Row norms of a (HW, 9C) bf16 patch matrix.
// ---------------------------------------------------------------------------
__global__ __launch_bounds__(256) void sl2_norm_kernel(
    const unsigned short* __restrict__ pm, float* __restrict__ nrm) {
    __shared__ float red[256];
    const int row = blockIdx.x;
    const unsigned short* __restrict__ r = pm + (size_t)row * K9C;
    float acc = 0.f;
    for (int k = threadIdx.x; k < K9C; k += 256) {
        float v = bf16_bits_to_f32(r[k]);
        acc += v * v;
    }
    red[threadIdx.x] = acc;
    __syncthreads();
    for (int st = 128; st > 0; st >>= 1) {
        if (threadIdx.x < st) red[threadIdx.x] += red[threadIdx.x + st];
        __syncthreads();
    }
    if (threadIdx.x == 0) nrm[row] = sqrtf(red[0]);
}

// ---------------------------------------------------------------------------
// 3) Init argmax table (reset every launch; deterministic under graph replay).
// ---------------------------------------------------------------------------
__global__ __launch_bounds__(256) void sl2_init_best_kernel(
    unsigned long long* __restrict__ best) {
    const int i = blockIdx.x * 256 + threadIdx.x;
    if (i < HW) best[i] = 0ull;
}

// ---------------------------------------------------------------------------
// 4) mask_sum = sum(mask), single block.
// ---------------------------------------------------------------------------
__global__ __launch_bounds__(256) void sl2_mask_sum_kernel(
    const float* __restrict__ mask, float* __restrict__ msum) {
    __shared__ float red[256];
    float acc = 0.f;
    for (int i = threadIdx.x; i < HW; i += 256) acc += mask[i];
    red[threadIdx.x] = acc;
    __syncthreads();
    for (int st = 128; st > 0; st >>= 1) {
        if (threadIdx.x < st) red[threadIdx.x] += red[threadIdx.x + st];
        __syncthreads();
    }
    if (threadIdx.x == 0) msum[0] = red[0];
}

// ---------------------------------------------------------------------------
// 5) Score GEMM + fused normalized argmax.
//    128x128 block / 256 threads (8 waves, 2x4): wave (wr,wc) owns rows
//    wr*64..+63 (4 m-tiles) x cols wc*32..+31 (2 n-tiles) -> 8 accumulators,
//    8 WMMAs per 32-deep K-step. LDS double-buffered, filled by async
//    global->LDS B128 copies (4 per thread per buffer).
// ---------------------------------------------------------------------------
__global__ __launch_bounds__(256) void sl2_score_argmax_kernel(
    const unsigned short* __restrict__ cp, const unsigned short* __restrict__ sp,
    const float* __restrict__ cn, const float* __restrict__ sn,
    unsigned long long* __restrict__ best) {
    __shared__ __align__(128) unsigned short As[2][128 * 32];   // 8 KB each
    __shared__ __align__(128) unsigned short Bs[2][128 * 32];

    const int tid  = threadIdx.x;
    const int wave = tid >> 5;
    const int lane = tid & 31;
    const int wr   = wave >> 2;           // 0..1
    const int wc   = wave & 3;            // 0..3
    const int m0   = blockIdx.y * 128;
    const int n0   = blockIdx.x * 128;

    // loader: 512 16B chunks per matrix per K-step; thread takes chunk tid, tid+256
    const int r0 = tid >> 2;                    // chunk row, q=0
    const int s0 = (tid & 3) << 3;              // chunk seg (shorts)
    const int r1 = (tid + 256) >> 2;
    const int s1 = s0;                          // (ch+256)&3 == ch&3

    // fragment offsets (shorts): contiguous k order used for A and B, lanes 0-15 K0..15, 16-31 K16..31
    const int kb   = (lane >> 4) << 4;
    const int l15  = lane & 15;
    int aOff[4], bOff[2];
    #pragma unroll
    for (int mt = 0; mt < 4; ++mt) aOff[mt] = ((wr << 6) + (mt << 4) + l15) * 32 + kb;
    #pragma unroll
    for (int bt = 0; bt < 2; ++bt) bOff[bt] = ((wc << 5) + (bt << 4) + l15) * 32 + kb;

    v8f acc[4][2];
    #pragma unroll
    for (int mt = 0; mt < 4; ++mt)
        #pragma unroll
        for (int bt = 0; bt < 2; ++bt)
            acc[mt][bt] = (v8f){0.f, 0.f, 0.f, 0.f, 0.f, 0.f, 0.f, 0.f};

    #define SL2_ISSUE(buf, k0)                                                          \
        do {                                                                            \
            sl2_copy_b128(&cp[(size_t)(m0 + r0) * K9C + (k0) + s0], &As[buf][r0 * 32 + s0]); \
            sl2_copy_b128(&sp[(size_t)(n0 + r0) * K9C + (k0) + s0], &Bs[buf][r0 * 32 + s0]); \
            sl2_copy_b128(&cp[(size_t)(m0 + r1) * K9C + (k0) + s1], &As[buf][r1 * 32 + s1]); \
            sl2_copy_b128(&sp[(size_t)(n0 + r1) * K9C + (k0) + s1], &Bs[buf][r1 * 32 + s1]); \
        } while (0)

    #define SL2_COMPUTE(buf)                                                            \
        do {                                                                            \
            v16bf a0 = *(const v16bf*)&As[buf][aOff[0]];                                \
            v16bf a1 = *(const v16bf*)&As[buf][aOff[1]];                                \
            v16bf a2 = *(const v16bf*)&As[buf][aOff[2]];                                \
            v16bf a3 = *(const v16bf*)&As[buf][aOff[3]];                                \
            v16bf b0 = *(const v16bf*)&Bs[buf][bOff[0]];                                \
            v16bf b1 = *(const v16bf*)&Bs[buf][bOff[1]];                                \
            acc[0][0] = __builtin_amdgcn_wmma_f32_16x16x32_bf16(false, a0, false, b0, (short)0, acc[0][0], false, false); \
            acc[1][0] = __builtin_amdgcn_wmma_f32_16x16x32_bf16(false, a1, false, b0, (short)0, acc[1][0], false, false); \
            acc[2][0] = __builtin_amdgcn_wmma_f32_16x16x32_bf16(false, a2, false, b0, (short)0, acc[2][0], false, false); \
            acc[3][0] = __builtin_amdgcn_wmma_f32_16x16x32_bf16(false, a3, false, b0, (short)0, acc[3][0], false, false); \
            acc[0][1] = __builtin_amdgcn_wmma_f32_16x16x32_bf16(false, a0, false, b1, (short)0, acc[0][1], false, false); \
            acc[1][1] = __builtin_amdgcn_wmma_f32_16x16x32_bf16(false, a1, false, b1, (short)0, acc[1][1], false, false); \
            acc[2][1] = __builtin_amdgcn_wmma_f32_16x16x32_bf16(false, a2, false, b1, (short)0, acc[2][1], false, false); \
            acc[3][1] = __builtin_amdgcn_wmma_f32_16x16x32_bf16(false, a3, false, b1, (short)0, acc[3][1], false, false); \
        } while (0)

    SL2_ISSUE(0, 0);
    int cur = 0;
    const int KT = K9C / 32;                    // 144
    for (int kt = 0; kt < KT - 1; ++kt) {
        SL2_ISSUE(cur ^ 1, (kt + 1) * 32);
        sl2_wait_async_4();                     // drain current buffer's copies
        __syncthreads();
        SL2_COMPUTE(cur);
        __syncthreads();
        cur ^= 1;
    }
    sl2_wait_async_0();
    __syncthreads();
    SL2_COMPUTE(cur);

    // C/D layout: VGPR r of lane l -> M = (l>>4)*8 + r, N = l&15.
    const int mbase = m0 + (wr << 6) + ((lane >> 4) << 3);
    const int nbase = n0 + (wc << 5) + l15;
    #pragma unroll
    for (int mt = 0; mt < 4; ++mt) {
        #pragma unroll
        for (int r = 0; r < 8; ++r) {
            const int m  = mbase + (mt << 4) + r;
            const float cm = cn[m];
            #pragma unroll
            for (int bt = 0; bt < 2; ++bt) {
                const int n = nbase + (bt << 4);
                float sc = acc[mt][bt][r] / (cm * sn[n] + EPSF);
                unsigned int fb  = __float_as_uint(sc);
                unsigned int key = (fb & 0x80000000u) ? ~fb : (fb | 0x80000000u);
                atomicMax(&best[m], ((unsigned long long)key << 32) | (unsigned int)n);
            }
        }
    }
    #undef SL2_ISSUE
    #undef SL2_COMPUTE
}

// ---------------------------------------------------------------------------
// 6) Gather matched style pixels + mask; also masked input fm. -> bf16 (C,HW)
// ---------------------------------------------------------------------------
__global__ __launch_bounds__(256) void sl2_gather_mask_kernel(
    const float* __restrict__ style, const float* __restrict__ input,
    const float* __restrict__ mask, const unsigned long long* __restrict__ best,
    unsigned short* __restrict__ Fs, unsigned short* __restrict__ Fi) {
    const int t = blockIdx.x * 256 + threadIdx.x;   // over 512*4096
    const int c = t >> 12;
    const int p = t & 4095;
    const int idx = (int)(best[p] & 0xFFFFFFFFull);
    const float mk = mask[p];
    Fs[t] = f32_to_bf16_bits(style[(size_t)c * HW + idx] * mk);
    Fi[t] = f32_to_bf16_bits(input[t] * mk);
}

// ---------------------------------------------------------------------------
// 7) Masked Gram: G = (F @ F^T) / mask_sum. F (512,4096) bf16 row-major, so
//    B = F^T tiles load with the same row-major pattern as A.
// ---------------------------------------------------------------------------
__global__ __launch_bounds__(256) void sl2_gram_kernel(
    const unsigned short* __restrict__ F, const float* __restrict__ msum,
    float* __restrict__ gram) {
    __shared__ __align__(64) unsigned short As[64][32];
    __shared__ __align__(64) unsigned short Bs[64][32];

    const int tid  = threadIdx.x;
    const int wave = tid >> 5;
    const int lane = tid & 31;
    const int wr   = wave >> 1;
    const int wc   = wave & 1;
    const int m0   = blockIdx.y * 64;
    const int n0   = blockIdx.x * 64;

    const int ldRow = tid >> 2;
    const int ldCol = (tid & 3) << 3;

    v8f acc0 = {0.f, 0.f, 0.f, 0.f, 0.f, 0.f, 0.f, 0.f};
    v8f acc1 = {0.f, 0.f, 0.f, 0.f, 0.f, 0.f, 0.f, 0.f};

    const int mrow  = wr * 16 + (lane & 15);
    const int brow0 = wc * 32 + (lane & 15);
    const int kb    = (lane >> 4) << 4;

    for (int k0 = 0; k0 < HW; k0 += 32) {
        *(uint4*)&As[ldRow][ldCol] =
            *(const uint4*)&F[(size_t)(m0 + ldRow) * HW + k0 + ldCol];
        *(uint4*)&Bs[ldRow][ldCol] =
            *(const uint4*)&F[(size_t)(n0 + ldRow) * HW + k0 + ldCol];
        __syncthreads();

        v16bf a  = *(const v16bf*)&As[mrow][kb];
        v16bf b0 = *(const v16bf*)&Bs[brow0][kb];
        v16bf b1 = *(const v16bf*)&Bs[brow0 + 16][kb];
        acc0 = __builtin_amdgcn_wmma_f32_16x16x32_bf16(
                   false, a, false, b0, (short)0, acc0, false, false);
        acc1 = __builtin_amdgcn_wmma_f32_16x16x32_bf16(
                   false, a, false, b1, (short)0, acc1, false, false);
        __syncthreads();
    }

    const float inv = 1.0f / msum[0];
    const int mbase = m0 + wr * 16 + ((lane >> 4) << 3);
    const int nbase = n0 + wc * 32 + (lane & 15);
    #pragma unroll
    for (int r = 0; r < 8; ++r) {
        gram[(size_t)(mbase + r) * CCH + nbase]      = acc0[r] * inv;
        gram[(size_t)(mbase + r) * CCH + nbase + 16] = acc1[r] * inv;
    }
}

// ---------------------------------------------------------------------------
// 8) loss = mean((G - target)^2) * 100, single block.
// ---------------------------------------------------------------------------
__global__ __launch_bounds__(256) void sl2_loss_kernel(
    const float* __restrict__ gt, const float* __restrict__ gg,
    float* __restrict__ out) {
    __shared__ float red[256];
    float acc = 0.f;
    for (int i = threadIdx.x; i < CCH * CCH; i += 256) {
        float d = gg[i] - gt[i];
        acc += d * d;
    }
    red[threadIdx.x] = acc;
    __syncthreads();
    for (int st = 128; st > 0; st >>= 1) {
        if (threadIdx.x < st) red[threadIdx.x] += red[threadIdx.x + st];
        __syncthreads();
    }
    if (threadIdx.x == 0)
        out[0] = (red[0] / (float)(CCH * CCH)) * 100.0f;
}

// ---------------------------------------------------------------------------
// Launch
// ---------------------------------------------------------------------------
extern "C" void kernel_launch(void* const* d_in, const int* in_sizes, int n_in,
                              void* d_out, int out_size, void* d_ws, size_t ws_size,
                              hipStream_t stream) {
    const float* content = (const float*)d_in[0];   // (1,512,64,64)
    const float* style   = (const float*)d_in[1];   // (1,512,64,64)
    const float* input   = (const float*)d_in[2];   // (1,512,64,64)
    const float* mask    = (const float*)d_in[3];   // (1,1,64,64)
    float* out = (float*)d_out;

    // Workspace carve-up (all offsets 256B-aligned), ~86.1 MB total.
    char* ws = (char*)d_ws;
    unsigned short*     cp   = (unsigned short*)(ws + 0);                   // 37,748,736
    unsigned short*     sp   = (unsigned short*)(ws + 37748736u);           // 37,748,736
    float*              cn   = (float*)(ws + 75497472u);                    //     16,384
    float*              sn   = (float*)(ws + 75513856u);                    //     16,384
    unsigned long long* best = (unsigned long long*)(ws + 75530240u);       //     32,768
    unsigned short*     Fs   = (unsigned short*)(ws + 75563008u);           //  4,194,304
    unsigned short*     Fi   = (unsigned short*)(ws + 79757312u);           //  4,194,304
    float*              gt   = (float*)(ws + 83951616u);                    //  1,048,576
    float*              gg   = (float*)(ws + 85000192u);                    //  1,048,576
    float*              msum = (float*)(ws + 86048768u);                    //          4

    // 1) bf16 patch matrices (coalesced via LDS transpose)
    sl2_patches_kernel<<<9 * 16 * 128, 256, 0, stream>>>(content, cp);
    sl2_patches_kernel<<<9 * 16 * 128, 256, 0, stream>>>(style, sp);
    // 2) row norms
    sl2_norm_kernel<<<HW, 256, 0, stream>>>(cp, cn);
    sl2_norm_kernel<<<HW, 256, 0, stream>>>(sp, sn);
    // 3) reset argmax table, 4) mask sum
    sl2_init_best_kernel<<<HW / 256, 256, 0, stream>>>(best);
    sl2_mask_sum_kernel<<<1, 256, 0, stream>>>(mask, msum);
    // 5) 4096x4096x4608 WMMA GEMM + fused normalized argmax
    sl2_score_argmax_kernel<<<dim3(HW / 128, HW / 128), 256, 0, stream>>>(
        cp, sp, cn, sn, best);
    // 6) gather matched style + masking (bf16)
    sl2_gather_mask_kernel<<<(CCH * HW) / 256, 256, 0, stream>>>(
        style, input, mask, best, Fs, Fi);
    // 7) masked Grams via WMMA
    sl2_gram_kernel<<<dim3(CCH / 64, CCH / 64), 256, 0, stream>>>(Fs, msum, gt);
    sl2_gram_kernel<<<dim3(CCH / 64, CCH / 64), 256, 0, stream>>>(Fi, msum, gg);
    // 8) MSE * 100
    sl2_loss_kernel<<<1, 256, 0, stream>>>(gt, gg, out);
}
